// ScannedRNN_37589553775139
// MI455X (gfx1250) — compile-verified, matrix operand order checked by reading, not run on hardware
//
#include <hip/hip_runtime.h>
#include <hip/hip_bf16.h>
#include <math.h>

// Problem constants (from reference)
#define T_DIM 512
#define B_DIM 128
#define D_DIM 512
#define H_DIM 512
#define G_DIM (3 * H_DIM)   // 1536
#define STEP_BLOCKS 32      // persistent recurrence blocks (co-resident)

typedef __attribute__((ext_vector_type(16))) __bf16 bf16x16;
typedef __attribute__((ext_vector_type(8)))  float  v8f;

// Load a 16-bit WMMA fragment half for this lane: two contiguous 16-byte
// chunks at [base, base+8) and [base+16, base+24) (bf16 elements).
// Matches ISA 16-bit A/B layout: lane<16 holds K=k0..k0+7 and k0+16..k0+23.
__device__ __forceinline__ bf16x16 load_frag16(const __bf16* base) {
    union { uint4 u[2]; bf16x16 v; } u;
    u.u[0] = *(const uint4*)(base);
    u.u[1] = *(const uint4*)(base + 16);
    return u.v;
}

__device__ __forceinline__ v8f wmma_bf16(bf16x16 a, bf16x16 b, v8f c) {
    // (neg_a, A, neg_b, B, c_mod, C, reuse_a, reuse_b)
    return __builtin_amdgcn_wmma_f32_16x16x32_bf16(false, a, false, b,
                                                   (short)0, c, false, false);
}

__device__ __forceinline__ float sigmoidf_fast(float x) {
    return 1.0f / (1.0f + __expf(-x));
}

// ---------------------------------------------------------------------------
// Kernel 1: one-time prep. bf16 copies of ins; transposed bf16 Wi/Wh; h init;
// zero the inter-block barrier state (re-zeroed every call => graph-replay
// deterministic). WiT[n][k] = Wi[k][n] so B-fragment rows are contiguous.
// ---------------------------------------------------------------------------
__global__ __launch_bounds__(256) void prep_kernel(
    const float* __restrict__ ins, const float* __restrict__ Wi,
    const float* __restrict__ Wh,  const float* __restrict__ h0,
    __bf16* __restrict__ insb, __bf16* __restrict__ WiT,
    __bf16* __restrict__ WhT, float* __restrict__ h32_0,
    __bf16* __restrict__ hb_0, unsigned* __restrict__ bar)
{
    long long i = (long long)blockIdx.x * blockDim.x + threadIdx.x;
    const long long NINS = (long long)T_DIM * B_DIM * D_DIM;
    if (i < NINS) insb[i] = (__bf16)ins[i];
    if (i < (long long)G_DIM * D_DIM) {
        int n = (int)(i / D_DIM), k = (int)(i % D_DIM);
        WiT[i] = (__bf16)Wi[(long long)k * G_DIM + n];
    }
    if (i < (long long)G_DIM * H_DIM) {
        int n = (int)(i / H_DIM), k = (int)(i % H_DIM);
        WhT[i] = (__bf16)Wh[(long long)k * G_DIM + n];
    }
    if (i < (long long)B_DIM * H_DIM) {
        float v = h0[i];
        h32_0[i] = v;
        hb_0[i]  = (__bf16)v;
    }
    if (i < 2) bar[i] = 0u;   // bar[0] = arrive count, bar[1] = generation
}

// ---------------------------------------------------------------------------
// Kernel 2: Gi[t*B+b][n] = ins[t,b,:] @ Wi[:,n] + bi[n], one big GEMM.
// M = T*B = 65536, K = D = 512, N = 3H = 1536.
// 32x32 register blocking per wave: 2 A frags x 2 B frags -> 4 accumulators,
// 4 WMMAs per 32-wide K chunk. Halves L2 fragment traffic per output vs 16x16
// and gives 4 independent WMMAs per chunk for XDL latency hiding.
// grid = (48 N-tiles, 256 groups of 8 M-tiles), block = 256 (8 waves).
// ---------------------------------------------------------------------------
__global__ __launch_bounds__(256) void gi_gemm_kernel(
    const __bf16* __restrict__ insb, const __bf16* __restrict__ WiT,
    const float* __restrict__ bi, float* __restrict__ Gi)
{
    const int wave = threadIdx.x >> 5;
    const int lane = threadIdx.x & 31;
    const int n0   = blockIdx.x * 32;
    const long long m0 = ((long long)blockIdx.y * 8 + wave) * 32;

    const int khalf = (lane < 16) ? 0 : 8;
    const __bf16* arow0 = insb + (m0 + (lane & 15)) * D_DIM + khalf;
    const __bf16* arow1 = arow0 + 16 * D_DIM;
    const __bf16* brow0 = WiT + (long long)(n0 + (lane & 15)) * D_DIM + khalf;
    const __bf16* brow1 = brow0 + 16 * D_DIM;

    v8f c00 = {}, c01 = {}, c10 = {}, c11 = {};
#pragma unroll 4
    for (int k = 0; k < D_DIM; k += 32) {
        __builtin_prefetch(arow0 + k + 128, 0, 0);   // global_prefetch_b8
        __builtin_prefetch(arow1 + k + 128, 0, 0);
        bf16x16 a0 = load_frag16(arow0 + k);
        bf16x16 a1 = load_frag16(arow1 + k);
        bf16x16 b0 = load_frag16(brow0 + k);
        bf16x16 b1 = load_frag16(brow1 + k);
        c00 = wmma_bf16(a0, b0, c00);
        c01 = wmma_bf16(a0, b1, c01);
        c10 = wmma_bf16(a1, b0, c10);
        c11 = wmma_bf16(a1, b1, c11);
    }

    // C/D layout: VGPR v <-> row v + (lane>=16 ? 8 : 0), lane&15 <-> column.
    const int rbase = (lane >= 16) ? 8 : 0;
    const int col0 = n0 + (lane & 15);
    const int col1 = col0 + 16;
    const float bi0 = bi[col0];
    const float bi1 = bi[col1];
#pragma unroll
    for (int v = 0; v < 8; ++v) {
        long long r0 = m0 + v + rbase;
        long long r1 = r0 + 16;
        Gi[r0 * G_DIM + col0] = c00[v] + bi0;
        Gi[r0 * G_DIM + col1] = c01[v] + bi1;
        Gi[r1 * G_DIM + col0] = c10[v] + bi0;
        Gi[r1 * G_DIM + col1] = c11[v] + bi1;
    }
}

// ---------------------------------------------------------------------------
// Kernel 3: PERSISTENT recurrence. One launch; T sequential steps separated by
// a device-scope atomic generation barrier (all STEP_BLOCKS blocks are
// co-resident on MI455X). Per step each wave owns a 16-row x 16-col strip of
// h and computes all three gate strips (r/z/n): 3 accumulators, 16 K chunks,
// 48 WMMAs, fused gate math. Wh^T (1.5 MB) and h (128 KB bf16) stay
// L2-resident; Gi gate rows (786 KB/step from HBM) are prefetched before the
// GEMM loop so the fetch overlaps the WMMAs.
// grid = 32 blocks x 256 threads (256 waves = 8 row-tiles x 32 col-strips).
// ---------------------------------------------------------------------------
__global__ __launch_bounds__(256) void gru_persistent_kernel(
    const float* __restrict__ Gi, const unsigned char* __restrict__ resets,
    const float* __restrict__ bhn, const __bf16* __restrict__ WhT,
    float* __restrict__ h32a, float* __restrict__ h32b,
    __bf16* __restrict__ hba, __bf16* __restrict__ hbb,
    float* __restrict__ ys, unsigned* __restrict__ bar)
{
    const int wave = (int)((blockIdx.x << 3) | (threadIdx.x >> 5)); // 0..255
    const int lane = threadIdx.x & 31;
    const int bt = wave & 7;     // row (batch) tile
    const int cs = wave >> 3;    // column strip within H
    const int j0 = cs * 16;
    const int colLane = j0 + (lane & 15);
    const int khalf = (lane < 16) ? 0 : 8;
    const int rbase = (lane >= 16) ? 8 : 0;
    const int b_for_A = bt * 16 + (lane & 15);

    const __bf16* brow_r = WhT + (long long)(0 * H_DIM + colLane) * H_DIM + khalf;
    const __bf16* brow_z = WhT + (long long)(1 * H_DIM + colLane) * H_DIM + khalf;
    const __bf16* brow_n = WhT + (long long)(2 * H_DIM + colLane) * H_DIM + khalf;
    const float bhnv = bhn[colLane];
    const bf16x16 zfrag = {};

    unsigned* cnt = bar + 0;
    unsigned* gen = bar + 1;

    for (int t = 0; t < T_DIM; ++t) {
        const float*  h32_in  = (t & 1) ? h32b : h32a;
        const __bf16* hb_in   = (t & 1) ? hbb  : hba;
        float*        h32_out = (t & 1) ? h32a : h32b;
        __bf16*       hb_out  = (t & 1) ? hba  : hbb;

        // Prefetch this step's Gi gate rows (HBM) to overlap with the WMMAs.
        const long long gbase = (long long)t * B_DIM * G_DIM;
#pragma unroll
        for (int v = 0; v < 8; ++v) {
            const float* g = Gi + gbase +
                (long long)(bt * 16 + v + rbase) * G_DIM + colLane;
            __builtin_prefetch(g, 0, 0);
            __builtin_prefetch(g + H_DIM, 0, 0);
            __builtin_prefetch(g + 2 * H_DIM, 0, 0);
        }

        // A operand: h rows (this lane holds batch row b_for_A); reset => 0.
        const bool rstA = resets[(long long)t * B_DIM + b_for_A] != 0;
        const __bf16* arow = hb_in + (long long)b_for_A * H_DIM + khalf;

        v8f acc_r = {}, acc_z = {}, acc_n = {};
#pragma unroll 4
        for (int k = 0; k < H_DIM; k += 32) {
            bf16x16 a = load_frag16(arow + k);
            if (rstA) a = zfrag;             // zero this A row (EXEC stays all-1)
            bf16x16 b0 = load_frag16(brow_r + k);
            bf16x16 b1 = load_frag16(brow_z + k);
            bf16x16 b2 = load_frag16(brow_n + k);
            acc_r = wmma_bf16(a, b0, acc_r);
            acc_z = wmma_bf16(a, b1, acc_z);
            acc_n = wmma_bf16(a, b2, acc_n);
        }

        // Fused gates on the 16x16 C/D tile.
#pragma unroll
        for (int v = 0; v < 8; ++v) {
            const int b = bt * 16 + v + rbase;
            const long long g = gbase + (long long)b * G_DIM;
            const float i_r = Gi[g + colLane];
            const float i_z = Gi[g + H_DIM + colLane];
            const float i_n = Gi[g + 2 * H_DIM + colLane];
            const bool rst = resets[(long long)t * B_DIM + b] != 0;
            const long long o = (long long)b * H_DIM + colLane;
            const float hprev = rst ? 0.0f : h32_in[o];
            const float r = sigmoidf_fast(i_r + acc_r[v]);
            const float z = sigmoidf_fast(i_z + acc_z[v]);
            const float n = tanhf(i_n + r * (acc_n[v] + bhnv));
            const float nh = (1.0f - z) * n + z * hprev;
            ys[(long long)t * B_DIM * H_DIM + o] = nh;
            h32_out[o] = nh;
            hb_out[o]  = (__bf16)nh;
        }

        // ---- device-scope generation barrier between steps ----
        __builtin_amdgcn_fence(__ATOMIC_RELEASE, "agent");  // publish h writes
        __syncthreads();
        if (threadIdx.x == 0) {
            const unsigned want = (unsigned)(t + 1);
            unsigned old = __hip_atomic_fetch_add(cnt, 1u, __ATOMIC_ACQ_REL,
                                                  __HIP_MEMORY_SCOPE_AGENT);
            if (old == STEP_BLOCKS - 1) {
                __hip_atomic_store(cnt, 0u, __ATOMIC_RELAXED,
                                   __HIP_MEMORY_SCOPE_AGENT);
                __hip_atomic_store(gen, want, __ATOMIC_RELEASE,
                                   __HIP_MEMORY_SCOPE_AGENT);
            } else {
                while (__hip_atomic_load(gen, __ATOMIC_ACQUIRE,
                                         __HIP_MEMORY_SCOPE_AGENT) < want) {
                    __builtin_amdgcn_s_sleep(2);
                }
            }
        }
        __syncthreads();
        __builtin_amdgcn_fence(__ATOMIC_ACQUIRE, "agent"); // invalidate stale L0
    }
}

// ---------------------------------------------------------------------------
// Kernel 4: copy final carry into d_out[0 .. B*H).
// ---------------------------------------------------------------------------
__global__ __launch_bounds__(256) void finalize_kernel(
    const float* __restrict__ h32_final, float* __restrict__ out)
{
    int i = blockIdx.x * blockDim.x + threadIdx.x;
    if (i < B_DIM * H_DIM) out[i] = h32_final[i];
}

extern "C" void kernel_launch(void* const* d_in, const int* in_sizes, int n_in,
                              void* d_out, int out_size, void* d_ws, size_t ws_size,
                              hipStream_t stream) {
    (void)in_sizes; (void)n_in; (void)out_size; (void)ws_size;
    const float*         ins    = (const float*)d_in[0];
    const unsigned char* resets = (const unsigned char*)d_in[1]; // jnp bool_
    const float*         h0     = (const float*)d_in[2];
    const float*         Wi     = (const float*)d_in[3];
    const float*         bi     = (const float*)d_in[4];
    const float*         Wh     = (const float*)d_in[5];
    const float*         bhn    = (const float*)d_in[6];
    float* out = (float*)d_out;                 // [B*H] h_final, then [T*B*H] ys
    float* ys  = out + (long long)B_DIM * H_DIM;

    // Workspace carve-up (256B aligned).
    char* ws = (char*)d_ws;
    size_t off = 0;
    auto take = [&](size_t bytes) -> char* {
        char* p = ws + off;
        off = (off + bytes + 255) & ~(size_t)255;
        return p;
    };
    __bf16*   insb = (__bf16*)take((size_t)T_DIM * B_DIM * D_DIM * 2);   // 67 MB
    __bf16*   WiT  = (__bf16*)take((size_t)G_DIM * D_DIM * 2);           // 1.5 MB
    __bf16*   WhT  = (__bf16*)take((size_t)G_DIM * H_DIM * 2);           // 1.5 MB
    float*    Gi   = (float*) take((size_t)T_DIM * B_DIM * G_DIM * 4);   // 402 MB
    float*    h32a = (float*) take((size_t)B_DIM * H_DIM * 4);
    float*    h32b = (float*) take((size_t)B_DIM * H_DIM * 4);
    __bf16*   hba  = (__bf16*)take((size_t)B_DIM * H_DIM * 2);
    __bf16*   hbb  = (__bf16*)take((size_t)B_DIM * H_DIM * 2);
    unsigned* bar  = (unsigned*)take(2 * sizeof(unsigned));

    // Phase 0: prep (bf16 copies, transposes, h init into buffer 0, barrier=0).
    {
        long long n = (long long)T_DIM * B_DIM * D_DIM;
        int blocks = (int)((n + 255) / 256);
        prep_kernel<<<blocks, 256, 0, stream>>>(ins, Wi, Wh, h0,
                                                insb, WiT, WhT, h32a, hba, bar);
    }

    // Phase 1: Gi = ins @ Wi + bi, one big bf16 WMMA GEMM (32x32 reg tiles).
    {
        dim3 grid(G_DIM / 32, (T_DIM * B_DIM) / (32 * 8));   // (48, 256)
        gi_gemm_kernel<<<grid, 256, 0, stream>>>(insb, WiT, bi, Gi);
    }

    // Phase 2: persistent recurrence, ONE launch, device barrier per step.
    gru_persistent_kernel<<<STEP_BLOCKS, 256, 0, stream>>>(
        Gi, resets, bhn, WhT, h32a, h32b, hba, hbb, ys, bar);

    // Phase 3: final carry. Last step (t = T-1, odd) wrote buffer index 0.
    finalize_kernel<<<(B_DIM * H_DIM + 255) / 256, 256, 0, stream>>>(
        h32a, out);
}
